// SwinBlock_85873576117060
// MI455X (gfx1250) — matrix operand order repaired and checked by reading, not compiled, not run on hardware
//
#include <hip/hip_runtime.h>
#include <hip/hip_bf16.h>
#include <math.h>

typedef _Float16 h16;
typedef __attribute__((ext_vector_type(16))) _Float16 v16h;
typedef __attribute__((ext_vector_type(8)))  _Float16 v8h;
typedef __attribute__((ext_vector_type(8)))  float    v8f;

#define WSZ   7
#define SHIFT 3
#define HEADS 6
#define HD    32
#define CDIM  192
#define DFF   768
#define NTOK  343          // 7^3 tokens per window
#define NPAD  352          // padded to 22*16
#define DSP   28           // spatial dim
#define WTOT  128          // total windows (2 batches * 64)
#define ROWS  43904        // 2*28^3  == WTOT*NTOK
#define MT    (ROWS/16)    // 2744 M-tiles

// ---------------------------------------------------------------- WMMA utils
__device__ __forceinline__ v8f wmma16(v16h a, v16h b, v8f c) {
  return __builtin_amdgcn_wmma_f32_16x16x32_f16(false, a, false, b, (short)0, c,
                                                false, false);
}
// A-tile 16x32 f16: lane&15 = row, lane>>4 selects K 0..15 vs 16..31
__device__ __forceinline__ v16h ldA(const h16* row, int kh) {
  v8h lo = *(const v8h*)(row + kh * 16);
  v8h hi = *(const v8h*)(row + 8 + kh * 16);
  v16h r;
#pragma unroll
  for (int i = 0; i < 8; ++i) { r[i] = lo[i]; r[i + 8] = hi[i]; }
  return r;
}
// B-tile 32x16 f16: lane&15 = col, lanes 0-15 hold K=0..15, lanes 16-31 K=16..31
__device__ __forceinline__ v16h ldB(const h16* col, int kh) {
  return *(const v16h*)(col + kh * 16);
}
__device__ __forceinline__ float wsum(float v) {
#pragma unroll
  for (int m = 16; m > 0; m >>= 1) v += __shfl_xor(v, m, 32);
  return v;
}
// shift-mask region of a global coordinate: [0,21) [21,25) [25,28)
__device__ __forceinline__ int reg3(int g) {
  return g < (DSP - WSZ) ? 0 : (g < (DSP - SHIFT) ? 1 : 2);
}
// generic LDS pointer -> 32-bit LDS byte address (flat LDS aperture keeps the
// workgroup-relative offset in addr[31:0], per ISA 10.2 aperture mapping)
__device__ __forceinline__ unsigned lds_addr32(const void* p) {
  return (unsigned)(uintptr_t)p;
}

// ------------------------------------------------------------- f32 -> f16
__global__ void k_cvt(const float* __restrict__ s, h16* __restrict__ d, int n) {
  int i = blockIdx.x * 256 + threadIdx.x;
  if (i < n) d[i] = (h16)s[i];
}

// -------------------------------------- LN1 + cyclic shift + window partition
__global__ __launch_bounds__(256) void k_ln1_part(const float* __restrict__ x,
                                                  const float* __restrict__ g,
                                                  const float* __restrict__ b,
                                                  h16* __restrict__ xw) {
  int row  = blockIdx.x * 8 + (threadIdx.x >> 5);   // window-ordered token row
  int lane = threadIdx.x & 31;
  int w = row / NTOK, tok = row % NTOK;
  int bb = w >> 6, wl = w & 63;
  int wd = wl >> 4, wh = (wl >> 2) & 3, ww = wl & 3;
  int tz = tok / 49, ty = (tok / 7) % 7, tx = tok % 7;
  int sd = (wd * WSZ + tz + SHIFT) % DSP;           // roll(-3): src = (s+3)%28
  int sh = (wh * WSZ + ty + SHIFT) % DSP;
  int sw = (ww * WSZ + tx + SHIFT) % DSP;
  const float* src = x + ((size_t)((bb * DSP + sd) * DSP + sh) * DSP + sw) * CDIM;
  float v[6]; float s = 0.f;
#pragma unroll
  for (int i = 0; i < 6; ++i) { v[i] = src[lane + 32 * i]; s += v[i]; }
  float mu = wsum(s) * (1.f / CDIM);
  float vv = 0.f;
#pragma unroll
  for (int i = 0; i < 6; ++i) { float d2 = v[i] - mu; vv += d2 * d2; }
  float rstd = rsqrtf(wsum(vv) * (1.f / CDIM) + 1e-5f);
  h16* dst = xw + (size_t)row * CDIM;
#pragma unroll
  for (int i = 0; i < 6; ++i) {
    int c = lane + 32 * i;
    dst[c] = (h16)((v[i] - mu) * rstd * g[c] + b[c]);
  }
}

// ------------------------------------------------------------- LN2
__global__ __launch_bounds__(256) void k_ln2(const float* __restrict__ xn,
                                             const float* __restrict__ g,
                                             const float* __restrict__ b,
                                             h16* __restrict__ h2) {
  int row  = blockIdx.x * 8 + (threadIdx.x >> 5);
  int lane = threadIdx.x & 31;
  const float* src = xn + (size_t)row * CDIM;
  float v[6]; float s = 0.f;
#pragma unroll
  for (int i = 0; i < 6; ++i) { v[i] = src[lane + 32 * i]; s += v[i]; }
  float mu = wsum(s) * (1.f / CDIM);
  float vv = 0.f;
#pragma unroll
  for (int i = 0; i < 6; ++i) { float d2 = v[i] - mu; vv += d2 * d2; }
  float rstd = rsqrtf(wsum(vv) * (1.f / CDIM) + 1e-5f);
  h16* dst = h2 + (size_t)row * CDIM;
#pragma unroll
  for (int i = 0; i < 6; ++i) {
    int c = lane + 32 * i;
    dst[c] = (h16)((v[i] - mu) * rstd * g[c] + b[c]);
  }
}

// ----------------------------------------------- WMMA mainloop: 16x32 out tile
__device__ __forceinline__ void gemm_16x32(const h16* __restrict__ A,
                                           const h16* __restrict__ W, int K,
                                           int mbase, int nbase, int lane,
                                           v8f& o0, v8f& o1) {
  int ln = lane & 15, kh = lane >> 4;
  const h16* ar = A + (size_t)(mbase + ln) * K;
  const h16* w0 = W + (size_t)(nbase + ln) * K;
  const h16* w1 = W + (size_t)(nbase + 16 + ln) * K;
  v8f a0 = {}; v8f a1 = {};
  for (int kk = 0; kk < K; kk += 32) {
    v16h a = ldA(ar + kk, kh);
    a0 = wmma16(a, ldB(w0 + kk, kh), a0);
    a1 = wmma16(a, ldB(w1 + kk, kh), a1);
  }
  o0 = a0; o1 = a1;
}

// ------------------------------------- QKV GEMM + scatter to q / k / v^T (f16)
__global__ __launch_bounds__(128) void k_gemm_qkv(const h16* __restrict__ A,
                                                  const h16* __restrict__ W,
                                                  const float* __restrict__ bias,
                                                  h16* __restrict__ qb,
                                                  h16* __restrict__ kb,
                                                  h16* __restrict__ vtb) {
  const int N32 = (3 * CDIM) / 32;                 // 18
  int tid = blockIdx.x * 4 + (threadIdx.x >> 5);
  if (tid >= MT * N32) return;
  int mt = tid / N32, nt = tid % N32;
  int lane = threadIdx.x & 31;
  v8f o0, o1;
  gemm_16x32(A, W, CDIM, mt * 16, nt * 32, lane, o0, o1);
  int ln = lane & 15, mhi = (lane >> 4) * 8;
#pragma unroll
  for (int r = 0; r < 8; ++r) {
    int row = mt * 16 + r + mhi;
    int w = row / NTOK, tok = row % NTOK;
#pragma unroll
    for (int t = 0; t < 2; ++t) {
      int c = nt * 32 + t * 16 + ln;
      float val = (t ? o1[r] : o0[r]) + bias[c];
      int sel = c / CDIM, ch = c % CDIM;
      int head = ch >> 5, hd = ch & 31;
      size_t whh = (size_t)(w * HEADS + head);
      if (sel == 0)
        qb[whh * NPAD * HD + (size_t)tok * HD + hd] = (h16)(val * 0.17677669529663687f);
      else if (sel == 1)
        kb[whh * NPAD * HD + (size_t)tok * HD + hd] = (h16)val;
      else
        vtb[whh * HD * NPAD + (size_t)hd * NPAD + tok] = (h16)val;
    }
  }
}

// ----------------------- fused attention per (window, head): QK^T+bias+mask,
// ----------------------- softmax, PxV — WMMA; K/V^T panels async-staged in LDS
__global__ __launch_bounds__(128) void k_attn(const h16* __restrict__ qb,
                                              const h16* __restrict__ kb,
                                              const h16* __restrict__ vtb,
                                              const float* __restrict__ rpb,
                                              h16* __restrict__ ao) {
  __shared__ __align__(32) h16 sS[4][16][NPAD];    // per-wave score strips 45KB
  __shared__ __align__(32) h16 sK[NPAD * HD];      // K panel   22.5KB
  __shared__ __align__(32) h16 sV[HD * NPAD];      // V^T panel 22.5KB
  int tidx = threadIdx.x;
  int lane = tidx & 31, wid = tidx >> 5;
  int blk = blockIdx.x;
  int head = blk % HEADS;
  int w = blk / HEADS;
  int wl = w & 63;
  int wd = wl >> 4, wh = (wl >> 2) & 3, ww = wl & 3;
  const h16* Q  = qb  + (size_t)(w * HEADS + head) * NPAD * HD;
  const h16* K  = kb  + (size_t)(w * HEADS + head) * NPAD * HD;
  const h16* Vt = vtb + (size_t)(w * HEADS + head) * HD * NPAD;
  int ln = lane & 15, kh = lane >> 4;

  // ---- stage K and V^T panels into LDS with gfx1250 async copies
  {
    const char* gk = (const char*)K;
    const char* gv = (const char*)Vt;
    unsigned lk = lds_addr32(&sK[0]);
    unsigned lv = lds_addr32(&sV[0]);
#pragma unroll
    for (int i = 0; i < 11; ++i) {                 // 11*128*16B = 22528B each
      unsigned o = (unsigned)(i * 128 + tidx) * 16u;
      unsigned long long ak = (unsigned long long)(uintptr_t)(gk + o);
      unsigned long long av = (unsigned long long)(uintptr_t)(gv + o);
      unsigned dk = lk + o, dv = lv + o;
      asm volatile("global_load_async_to_lds_b128 %0, %1, off"
                   :: "v"(dk), "v"(ak) : "memory");
      asm volatile("global_load_async_to_lds_b128 %0, %1, off"
                   :: "v"(dv), "v"(av) : "memory");
    }
    asm volatile("s_wait_asynccnt 0" ::: "memory");
  }
  __syncthreads();

  for (int rb = wid; rb < NPAD / 16; rb += 4) {
    // ---- S = Q x K^T for a 16-row strip, bias+mask applied on the way to LDS
    v16h aQ = ldA(Q + (size_t)(rb * 16 + ln) * HD, kh);
    for (int ct = 0; ct < NPAD / 16; ++ct) {
      v16h bK = *(const v16h*)&sK[(ct * 16 + ln) * HD + kh * 16];
      v8f c = {};
      c = wmma16(aQ, bK, c);
      int n = ct * 16 + ln;                         // key token (this lane's col)
      int jz = n / 49, jy = (n / 7) % 7, jx = n % 7;
      int rj = reg3(wd * WSZ + jz) * 9 + reg3(wh * WSZ + jy) * 3 + reg3(ww * WSZ + jx);
#pragma unroll
      for (int r = 0; r < 8; ++r) {
        int m = r + 8 * kh;
        int qt = rb * 16 + m;                       // query token
        float val;
        if (n >= NTOK || qt >= NTOK) {
          val = -60000.f;                           // padded -> exp underflows
        } else {
          int iz = qt / 49, iy = (qt / 7) % 7, ix = qt % 7;
          int bidx = ((iz - jz + 6) * 13 + (iy - jy + 6)) * 13 + (ix - jx + 6);
          int ri = reg3(wd * WSZ + iz) * 9 + reg3(wh * WSZ + iy) * 3 + reg3(ww * WSZ + ix);
          val = c[r] + rpb[bidx * HEADS + head] + ((ri == rj) ? 0.f : -100.f);
        }
        sS[wid][m][n] = (h16)val;
      }
    }
    asm volatile("s_wait_dscnt 0" ::: "memory");
    // ---- softmax: lane pair (row = ln, half = kh) scans 176 cols, v8h chunks
    int r = ln, qt = rb * 16 + r, j0 = kh * 176;
    float mx = -1e30f;
#pragma unroll
    for (int jj = 0; jj < 22; ++jj) {
      v8h c = *(const v8h*)&sS[wid][r][j0 + jj * 8];
#pragma unroll
      for (int e = 0; e < 8; ++e) mx = fmaxf(mx, (float)c[e]);
    }
    mx = fmaxf(mx, __shfl_xor(mx, 16, 32));
    float sum = 0.f;
#pragma unroll
    for (int jj = 0; jj < 22; ++jj) {
      v8h c = *(const v8h*)&sS[wid][r][j0 + jj * 8];
      v8h o;
#pragma unroll
      for (int e = 0; e < 8; ++e) {
        float ev = __expf((float)c[e] - mx);
        sum += ev;
        o[e] = (h16)ev;
      }
      *(v8h*)&sS[wid][r][j0 + jj * 8] = o;
    }
    sum += __shfl_xor(sum, 16, 32);
    float inv = (qt < NTOK) ? 1.f / sum : 0.f;      // padded query rows -> P=0
#pragma unroll
    for (int jj = 0; jj < 22; ++jj) {
      v8h c = *(const v8h*)&sS[wid][r][j0 + jj * 8];
#pragma unroll
      for (int e = 0; e < 8; ++e) c[e] = (h16)((float)c[e] * inv);
      *(v8h*)&sS[wid][r][j0 + jj * 8] = c;
    }
    asm volatile("s_wait_dscnt 0" ::: "memory");
    // ---- O = P x V   (A from LDS strip, B from LDS V^T panel)
#pragma unroll
    for (int nt2 = 0; nt2 < 2; ++nt2) {
      v8f acc = {};
      for (int kt = 0; kt < NPAD / 32; ++kt) {
        int kb2 = kt * 32;
        v8h lo = *(const v8h*)&sS[wid][ln][kb2 + kh * 16];
        v8h hi = *(const v8h*)&sS[wid][ln][kb2 + 8 + kh * 16];
        v16h aP;
#pragma unroll
        for (int i = 0; i < 8; ++i) { aP[i] = lo[i]; aP[i + 8] = hi[i]; }
        v16h bV = *(const v16h*)&sV[(nt2 * 16 + ln) * NPAD + kb2 + kh * 16];
        acc = wmma16(aP, bV, acc);
      }
#pragma unroll
      for (int r2 = 0; r2 < 8; ++r2) {
        int tok = rb * 16 + r2 + 8 * kh;
        if (tok < NTOK) {
          int ch = head * HD + nt2 * 16 + ln;
          ao[((size_t)w * NTOK + tok) * CDIM + ch] = (h16)acc[r2];
        }
      }
    }
  }
}

// ------------------- proj GEMM + window-reverse + roll(+3) + residual -> x_new
__global__ __launch_bounds__(128) void k_gemm_proj(const h16* __restrict__ A,
                                                   const h16* __restrict__ W,
                                                   const float* __restrict__ bias,
                                                   const float* __restrict__ x0,
                                                   float* __restrict__ xn) {
  const int N32 = CDIM / 32;                       // 6
  int tid = blockIdx.x * 4 + (threadIdx.x >> 5);
  if (tid >= MT * N32) return;
  int mt = tid / N32, nt = tid % N32;
  int lane = threadIdx.x & 31;
  v8f o0, o1;
  gemm_16x32(A, W, CDIM, mt * 16, nt * 32, lane, o0, o1);
  int ln = lane & 15, mhi = (lane >> 4) * 8;
#pragma unroll
  for (int r = 0; r < 8; ++r) {
    int row = mt * 16 + r + mhi;
    int w = row / NTOK, tok = row % NTOK;
    int bb = w >> 6, wl = w & 63;
    int wd = wl >> 4, wh = (wl >> 2) & 3, ww = wl & 3;
    int tz = tok / 49, ty = (tok / 7) % 7, tx = tok % 7;
    int od = (wd * WSZ + tz + SHIFT) % DSP;        // un-roll lands on source coord
    int oh = (wh * WSZ + ty + SHIFT) % DSP;
    int ow = (ww * WSZ + tx + SHIFT) % DSP;
    size_t base = ((size_t)((bb * DSP + od) * DSP + oh) * DSP + ow) * CDIM;
#pragma unroll
    for (int t = 0; t < 2; ++t) {
      int c = nt * 32 + t * 16 + ln;
      xn[base + c] = x0[base + c] + (t ? o1[r] : o0[r]) + bias[c];
    }
  }
}

// ----------------------------------------------- FC1 GEMM + exact GELU (f16 out)
__global__ __launch_bounds__(128) void k_gemm_fc1(const h16* __restrict__ A,
                                                  const h16* __restrict__ W,
                                                  const float* __restrict__ bias,
                                                  h16* __restrict__ ff) {
  const int N32 = DFF / 32;                        // 24
  int tid = blockIdx.x * 4 + (threadIdx.x >> 5);
  if (tid >= MT * N32) return;
  int mt = tid / N32, nt = tid % N32;
  int lane = threadIdx.x & 31;
  v8f o0, o1;
  gemm_16x32(A, W, CDIM, mt * 16, nt * 32, lane, o0, o1);
  int ln = lane & 15, mhi = (lane >> 4) * 8;
#pragma unroll
  for (int r = 0; r < 8; ++r) {
    int row = mt * 16 + r + mhi;
#pragma unroll
    for (int t = 0; t < 2; ++t) {
      int c = nt * 32 + t * 16 + ln;
      float v = (t ? o1[r] : o0[r]) + bias[c];
      float ge = 0.5f * v * (1.f + erff(v * 0.7071067811865475f));
      ff[(size_t)row * DFF + c] = (h16)ge;
    }
  }
}

// --------------------------------------- FC2 GEMM + residual -> f32 output
__global__ __launch_bounds__(128) void k_gemm_fc2(const h16* __restrict__ A,
                                                  const h16* __restrict__ W,
                                                  const float* __restrict__ bias,
                                                  const float* __restrict__ xn,
                                                  float* __restrict__ out) {
  const int N32 = CDIM / 32;                       // 6
  int tid = blockIdx.x * 4 + (threadIdx.x >> 5);
  if (tid >= MT * N32) return;
  int mt = tid / N32, nt = tid % N32;
  int lane = threadIdx.x & 31;
  v8f o0, o1;
  gemm_16x32(A, W, DFF, mt * 16, nt * 32, lane, o0, o1);
  int ln = lane & 15, mhi = (lane >> 4) * 8;
#pragma unroll
  for (int r = 0; r < 8; ++r) {
    size_t row = (size_t)(mt * 16 + r + mhi);
#pragma unroll
    for (int t = 0; t < 2; ++t) {
      int c = nt * 32 + t * 16 + ln;
      out[row * CDIM + c] = xn[row * CDIM + c] + (t ? o1[r] : o0[r]) + bias[c];
    }
  }
}

// --------------------------------------------------------------------- host
extern "C" void kernel_launch(void* const* d_in, const int* in_sizes, int n_in,
                              void* d_out, int out_size, void* d_ws, size_t ws_size,
                              hipStream_t stream) {
  (void)in_sizes; (void)n_in; (void)out_size; (void)ws_size;
  const float* x     = (const float*)d_in[0];
  const float* ln1g  = (const float*)d_in[1];
  const float* ln1b  = (const float*)d_in[2];
  const float* qkvw  = (const float*)d_in[3];
  const float* qkvb  = (const float*)d_in[4];
  const float* rpb   = (const float*)d_in[5];
  const float* projw = (const float*)d_in[6];
  const float* projb = (const float*)d_in[7];
  const float* ln2g  = (const float*)d_in[8];
  const float* ln2b  = (const float*)d_in[9];
  const float* fc1w  = (const float*)d_in[10];
  const float* fc1b  = (const float*)d_in[11];
  const float* fc2w  = (const float*)d_in[12];
  const float* fc2b  = (const float*)d_in[13];
  float* out = (float*)d_out;

  char* base = (char*)d_ws;
  size_t off = 0;
  auto take = [&](size_t bytes) {
    char* p = base + off;
    off = (off + bytes + 255) & ~(size_t)255;
    return p;
  };
  h16* wq  = (h16*)take((size_t)3 * CDIM * CDIM * 2);
  h16* wp  = (h16*)take((size_t)CDIM * CDIM * 2);
  h16* wf1 = (h16*)take((size_t)DFF * CDIM * 2);
  h16* wf2 = (h16*)take((size_t)CDIM * DFF * 2);
  h16* xw  = (h16*)take((size_t)ROWS * CDIM * 2);
  size_t qkvElems = (size_t)WTOT * HEADS * NPAD * HD;
  h16* qb  = (h16*)take(3 * qkvElems * 2);
  h16* kbuf = qb + qkvElems;
  h16* vtb  = kbuf + qkvElems;
  h16* ao  = (h16*)take((size_t)ROWS * CDIM * 2);
  float* xn = (float*)take((size_t)ROWS * CDIM * 4);
  h16* h2  = (h16*)take((size_t)ROWS * CDIM * 2);
  h16* ff  = (h16*)take((size_t)ROWS * DFF * 2);

  // weights -> f16
  k_cvt<<<(3 * CDIM * CDIM + 255) / 256, 256, 0, stream>>>(qkvw, wq, 3 * CDIM * CDIM);
  k_cvt<<<(CDIM * CDIM + 255) / 256, 256, 0, stream>>>(projw, wp, CDIM * CDIM);
  k_cvt<<<(DFF * CDIM + 255) / 256, 256, 0, stream>>>(fc1w, wf1, DFF * CDIM);
  k_cvt<<<(CDIM * DFF + 255) / 256, 256, 0, stream>>>(fc2w, wf2, CDIM * DFF);

  // LN1 + shift + window partition (ROWS/8 blocks of 8 waves)
  k_ln1_part<<<ROWS / 8, 256, 0, stream>>>(x, ln1g, ln1b, xw);

  // zero q/k/v^T padding (token 343..351 lanes)
  hipMemsetAsync(qb, 0, 3 * qkvElems * 2, stream);

  // QKV projection (WMMA) with head-scatter epilogue
  {
    int total = MT * ((3 * CDIM) / 32);
    k_gemm_qkv<<<(total + 3) / 4, 128, 0, stream>>>(xw, wq, qkvb, qb, kbuf, vtb);
  }
  // fused windowed attention: one block per (window, head)
  k_attn<<<WTOT * HEADS, 128, 0, stream>>>(qb, kbuf, vtb, rpb, ao);

  // proj + window reverse + roll-back + residual
  {
    int total = MT * (CDIM / 32);
    k_gemm_proj<<<(total + 3) / 4, 128, 0, stream>>>(ao, wp, projb, x, xn);
  }
  // LN2
  k_ln2<<<ROWS / 8, 256, 0, stream>>>(xn, ln2g, ln2b, h2);
  // MLP
  {
    int total = MT * (DFF / 32);
    k_gemm_fc1<<<(total + 3) / 4, 128, 0, stream>>>(h2, wf1, fc1b, ff);
  }
  {
    int total = MT * (CDIM / 32);
    k_gemm_fc2<<<(total + 3) / 4, 128, 0, stream>>>(ff, wf2, fc2b, xn, out);
  }
}
